// XAttention_35648228557162
// MI455X (gfx1250) — matrix operand order repaired and checked by reading, not compiled, not run on hardware
//
#include <hip/hip_runtime.h>

// ---------------- problem constants ----------------
#define DIMN   1024
#define HEADS  16
#define DHN    64
#define INNERN 4096
#define BQ     4
#define LQ     1024
#define LKV    2048
#define MQ     (BQ * LQ)    // 4096 query rows
#define MKV    (BQ * LKV)   // 8192 kv rows

typedef unsigned short u16;
typedef unsigned int   u32;
typedef __bf16 bf16_t;
typedef bf16_t v16bf __attribute__((ext_vector_type(16)));
typedef float  v8f   __attribute__((ext_vector_type(8)));
typedef int    v4i   __attribute__((ext_vector_type(4)));

// 16x32 (A) / 32x16 (B) bf16 fragment: 8 VGPRs = 32 bytes per lane
union FragBf { v16bf v; uint4 q[2]; };

static __device__ __forceinline__ u16 f2b(float f) {
    u32 u = __builtin_bit_cast(u32, f);
    u32 r = u + 0x7FFFu + ((u >> 16) & 1u);   // round-to-nearest-even
    return (u16)(r >> 16);
}
static __device__ __forceinline__ float b2f(u16 h) {
    u32 u = ((u32)h) << 16;
    return __builtin_bit_cast(float, u);
}
static __device__ __forceinline__ v8f wmma_bf(const FragBf& a, const FragBf& b, v8f c) {
    // D = A(16x32 bf16) x B(32x16 bf16) + C(16x16 f32)
    return __builtin_amdgcn_wmma_f32_16x16x32_bf16(false, a.v, false, b.v,
                                                   (short)0, c, false, false);
}

// ---------------- async global -> LDS staging (CDNA5) ----------------
#if defined(__gfx1250__) && __has_builtin(__builtin_amdgcn_global_load_async_to_lds_b128)
#define HAVE_ASYNC 1
#else
#define HAVE_ASYNC 0
#endif

// copy 16 bytes global -> LDS; async (ASYNCcnt) when available
static __device__ __forceinline__ void copy16(const u16* g, u16* l) {
#if HAVE_ASYNC
    __builtin_amdgcn_global_load_async_to_lds_b128((v4i*)g, (v4i*)l, 0, 0);
#else
    *(uint4*)l = *(const uint4*)g;
#endif
}

template <int N>
static __device__ __forceinline__ void wait_async() {
#if HAVE_ASYNC
#if __has_builtin(__builtin_amdgcn_s_wait_asynccnt)
    __builtin_amdgcn_s_wait_asynccnt((short)N);
#else
    asm volatile("s_wait_asynccnt %0" :: "i"(N) : "memory");
#endif
#endif
}

// ---------------- prep kernels ----------------

// out[C,R] (bf16) = transpose(in[R,C] f32); R,C multiples of 32
__global__ void __launch_bounds__(256) transpose_to_bf16(
    const float* __restrict__ in, u16* __restrict__ out, int R, int C)
{
    __shared__ float t[32][33];
    const int bx = blockIdx.x * 32;   // C
    const int by = blockIdx.y * 32;   // R
    const int tx = threadIdx.x;       // 0..31
    const int ty = threadIdx.y;       // 0..7
#pragma unroll
    for (int i = 0; i < 32; i += 8)
        t[ty + i][tx] = in[(size_t)(by + ty + i) * C + (bx + tx)];
    __syncthreads();
#pragma unroll
    for (int i = 0; i < 32; i += 8)
        out[(size_t)(bx + ty + i) * R + (by + tx)] = f2b(t[tx][ty + i]);
}

// row-wise LayerNorm over D=1024, fp32 in -> bf16 out; one block (256 thr) per row
__global__ void __launch_bounds__(256) layernorm_to_bf16(
    const float* __restrict__ x, const float* __restrict__ g,
    const float* __restrict__ be, u16* __restrict__ out)
{
    const int row = blockIdx.x;
    const int tid = threadIdx.x;
    const float4 v = *(const float4*)(x + (size_t)row * DIMN + tid * 4);
    float s  = v.x + v.y + v.z + v.w;
    float s2 = v.x * v.x + v.y * v.y + v.z * v.z + v.w * v.w;
#pragma unroll
    for (int d = 1; d < 32; d <<= 1) {
        s  += __shfl_xor(s,  d, 32);
        s2 += __shfl_xor(s2, d, 32);
    }
    __shared__ float rs[8], rs2[8];
    if ((tid & 31) == 0) { rs[tid >> 5] = s; rs2[tid >> 5] = s2; }
    __syncthreads();
    float S = 0.f, S2 = 0.f;
#pragma unroll
    for (int i = 0; i < 8; ++i) { S += rs[i]; S2 += rs2[i]; }
    const float mean = S * (1.0f / DIMN);
    const float var  = S2 * (1.0f / DIMN) - mean * mean;
    const float rstd = rsqrtf(var + 1e-5f);
    const float4 gv = *(const float4*)(g  + tid * 4);
    const float4 bv = *(const float4*)(be + tid * 4);
    uint2 pk;
    pk.x = (u32)f2b((v.x - mean) * rstd * gv.x + bv.x) |
           ((u32)f2b((v.y - mean) * rstd * gv.y + bv.y) << 16);
    pk.y = (u32)f2b((v.z - mean) * rstd * gv.z + bv.z) |
           ((u32)f2b((v.w - mean) * rstd * gv.w + bv.w) << 16);
    *(uint2*)(out + (size_t)row * DIMN + tid * 4) = pk;
}

// kvp = kv + pos_embed (broadcast over batch), fp32 -> bf16
__global__ void __launch_bounds__(256) kv_plus_pos_bf16(
    const float* __restrict__ kv, const float* __restrict__ pos, u16* __restrict__ out)
{
    const size_t i4   = (size_t)blockIdx.x * 256 + threadIdx.x;  // float4 index
    const size_t base = i4 * 4;
    const int col  = (int)(base & (DIMN - 1));
    const int row  = (int)(base >> 10);
    const int prow = row & (LKV - 1);
    const float4 a = *(const float4*)(kv + base);
    const float4 p = *(const float4*)(pos + (size_t)prow * DIMN + col);
    uint2 pk;
    pk.x = (u32)f2b(a.x + p.x) | ((u32)f2b(a.y + p.y) << 16);
    pk.y = (u32)f2b(a.z + p.z) | ((u32)f2b(a.w + p.w) << 16);
    *(uint2*)(out + base) = pk;
}

// ---------------- WMMA GEMM, LDS double-buffered, fused epilogues ----------------
// C[M,N] f32 = A_bf16[M,K](lda) x Bt_bf16[N,K](ldb)  (Bt = B transposed)
// Block: 256 threads = 8 waves (4 along M x 2 along N), block tile 256x128,
// wave tile 64x64, K-step 32 staged through LDS (2 x (16KB A + 8KB B)).
// EPI: 1 = store bf16(Cb)
//      2 = +resid -> store f32(Cf) and bf16(Cb)          (attn_out)
//      3 = exact GELU -> store bf16(Cb)                  (h@W1)
//      4 = * aux_bf16 -> store bf16(Cb)                  (GEGLU gate)
//      5 = +resid -> store f32(Cf)                       (final out)
//      6 = KV split: n<DIMN -> keys(Cb); else V transposed per head (Cb2)
template <int EPI>
__global__ void __launch_bounds__(256) gemm_bf16(
    const u16* __restrict__ A, int lda,
    const u16* __restrict__ Bt, int ldb,
    float* __restrict__ Cf, u16* __restrict__ Cb, u16* __restrict__ Cb2,
    const float* __restrict__ resid, const u16* __restrict__ aux,
    int M, int N, int K)
{
    __shared__ u16 lsA[2][256 * 32];   // [row within block][k within step]
    __shared__ u16 lsB[2][128 * 32];
    const int tid  = threadIdx.x;
    const int lane = tid & 31;
    const int wave = tid >> 5;
    const int hf   = lane >> 4;     // K-half selector (A/B lane layout)
    const int l15  = lane & 15;
    const int wm   = wave >> 1;     // 0..3
    const int wn   = wave & 1;      // 0..1
    const int bm   = blockIdx.y * 256;
    const int bn   = blockIdx.x * 128;
    const int m0   = bm + wm * 64;
    const int n0   = bn + wn * 64;

    v8f acc[4][4];
#pragma unroll
    for (int i = 0; i < 4; ++i)
#pragma unroll
        for (int j = 0; j < 4; ++j)
#pragma unroll
            for (int e = 0; e < 8; ++e) acc[i][j][e] = 0.0f;

    // stage one K-step: A 1024 x16B chunks, B 512 x16B chunks (6 copies/thread)
    auto issue = [&](int k0, int buf) {
#pragma unroll
        for (int i = 0; i < 4; ++i) {
            const int c = i * 256 + tid;
            const int row = c >> 2, part = c & 3;     // 4 chunks per 64B row
            copy16(A + (size_t)(bm + row) * lda + k0 + part * 8,
                   &lsA[buf][row * 32 + part * 8]);
        }
#pragma unroll
        for (int i = 0; i < 2; ++i) {
            const int c = i * 256 + tid;
            const int row = c >> 2, part = c & 3;
            copy16(Bt + (size_t)(bn + row) * ldb + k0 + part * 8,
                   &lsB[buf][row * 32 + part * 8]);
        }
    };

    int buf = 0;
    issue(0, 0);
    for (int k0 = 0; k0 < K; k0 += 32) {
        const bool more = (k0 + 32 < K);
        if (more) issue(k0 + 32, buf ^ 1);
        if (more) wait_async<6>(); else wait_async<0>();
        __syncthreads();

        FragBf a[4], b[4];
#pragma unroll
        for (int i = 0; i < 4; ++i) {
            const u16* ap = &lsA[buf][(wm * 64 + i * 16 + l15) * 32 + hf * 8];
            a[i].q[0] = *(const uint4*)(ap);
            a[i].q[1] = *(const uint4*)(ap + 16);
        }
#pragma unroll
        for (int j = 0; j < 4; ++j) {
            const u16* bp = &lsB[buf][(wn * 64 + j * 16 + l15) * 32 + hf * 8];
            b[j].q[0] = *(const uint4*)(bp);
            b[j].q[1] = *(const uint4*)(bp + 16);
        }
#pragma unroll
        for (int i = 0; i < 4; ++i)
#pragma unroll
            for (int j = 0; j < 4; ++j)
                acc[i][j] = wmma_bf(a[i], b[j], acc[i][j]);

        __syncthreads();   // all waves done with buf before it is re-targeted
        buf ^= 1;
    }

    // epilogue: C-layout — VGPR e holds row (hf*8 + e), col = l15 (per 16x16 tile)
#pragma unroll
    for (int i = 0; i < 4; ++i)
#pragma unroll
        for (int e = 0; e < 8; ++e) {
            const int gm = m0 + i * 16 + hf * 8 + e;
#pragma unroll
            for (int j = 0; j < 4; ++j) {
                const int gn = n0 + j * 16 + l15;
                const float vv = acc[i][j][e];
                const size_t idx = (size_t)gm * N + gn;
                if (EPI == 1) {
                    Cb[idx] = f2b(vv);
                } else if (EPI == 2) {
                    const float r = vv + resid[idx];
                    Cf[idx] = r; Cb[idx] = f2b(r);
                } else if (EPI == 3) {
                    const float ge = 0.5f * vv * (1.0f + erff(vv * 0.70710678118f));
                    Cb[idx] = f2b(ge);
                } else if (EPI == 4) {
                    Cb[idx] = f2b(vv * b2f(aux[idx]));
                } else if (EPI == 5) {
                    Cf[idx] = vv + resid[idx];
                } else if (EPI == 6) {
                    const int bb = gm >> 11, lkv = gm & (LKV - 1);
                    if (gn < DIMN) {
                        Cb[(size_t)gm * DIMN + gn] = f2b(vv);        // keys [B*Lkv, D]
                    } else {
                        const int dhg = gn - DIMN, hh = dhg >> 6, dh = dhg & 63;
                        // Vt [B*H*DH, Lkv] : row = ((b*H+h)*64+dh)
                        Cb2[((size_t)((bb * HEADS + hh) * DHN + dh) << 11) + lkv] = f2b(vv);
                    }
                }
            }
        }
}

// ---------------- fused flash attention (LDS double-buffered K/V) ----------------
// grid (LQ/128, HEADS, B), 256 threads = 8 waves; each wave owns 16 q-rows.
// All 8 waves share the same (b,h) K/V stream -> stage 64-wide chunks once per block.
__global__ void __launch_bounds__(256) flash_attn(
    const u16* __restrict__ Q,    // [B*Lq, D] bf16 (queries)
    const u16* __restrict__ Kb,   // [B*Lkv, D] bf16 (keys)
    const u16* __restrict__ Vt,   // [B*H*DH, Lkv] bf16 (values, transposed per head)
    const unsigned char* __restrict__ amask,  // [B, Lkv] bool
    u16* __restrict__ ctx)        // [B*Lq, D] bf16
{
    __shared__ u16 lsK[2][64 * 64];           // [kv row][dh], 8KB per buffer
    __shared__ u16 lsV[2][64 * 64];           // [dh row][kv], 8KB per buffer
    __shared__ u16 pbuf[8][16][64];           // per-wave P staging (2 KB each)
    const int tid  = threadIdx.x;
    const int lane = tid & 31;
    const int wave = tid >> 5;
    const int hf   = lane >> 4;
    const int l15  = lane & 15;
    const int bb   = blockIdx.z;
    const int hh   = blockIdx.y;
    const int m0   = blockIdx.x * 128 + wave * 16;   // q row within Lq

    // Q fragments (K-dim = dh = 64 -> two 16x32 A-frags), kept in registers
    FragBf aq0, aq1;
    {
        const u16* qrow = Q + (size_t)(bb * LQ + m0 + l15) * DIMN + hh * DHN;
        const int kb = hf * 8;
        aq0.q[0] = *(const uint4*)(qrow + kb);
        aq0.q[1] = *(const uint4*)(qrow + kb + 16);
        aq1.q[0] = *(const uint4*)(qrow + 32 + kb);
        aq1.q[1] = *(const uint4*)(qrow + 32 + kb + 16);
    }

    v8f O[4];
    float mrow[8], lrow[8];
#pragma unroll
    for (int od = 0; od < 4; ++od)
#pragma unroll
        for (int e = 0; e < 8; ++e) O[od][e] = 0.0f;
#pragma unroll
    for (int e = 0; e < 8; ++e) { mrow[e] = -3.0e38f; lrow[e] = 0.0f; }

    // stage one kv chunk: K 512 + V 512 x16B chunks (4 copies/thread)
    auto issue = [&](int nk, int buf) {
#pragma unroll
        for (int i = 0; i < 2; ++i) {
            const int c = i * 256 + tid;
            const int row = c >> 3, part = c & 7;   // 8 chunks per 128B row
            copy16(Kb + (size_t)(bb * LKV + nk + row) * DIMN + hh * DHN + part * 8,
                   &lsK[buf][row * 64 + part * 8]);
        }
#pragma unroll
        for (int i = 0; i < 2; ++i) {
            const int c = i * 256 + tid;
            const int row = c >> 3, part = c & 7;
            copy16(Vt + (size_t)((bb * HEADS + hh) * DHN + row) * LKV + nk + part * 8,
                   &lsV[buf][row * 64 + part * 8]);
        }
    };

    int buf = 0;
    issue(0, 0);
    for (int nk = 0; nk < LKV; nk += 64) {
        const bool more = (nk + 64 < LKV);
        if (more) issue(nk + 64, buf ^ 1);
        if (more) wait_async<4>(); else wait_async<0>();
        __syncthreads();

        // ---- S = Q @ K^T for a 16x64 chunk (4 n-subtiles) ----
        v8f S[4];
        float madd[4];
#pragma unroll
        for (int t = 0; t < 4; ++t) {
            const u16* krow = &lsK[buf][(t * 16 + l15) * 64];
            FragBf bk0, bk1;
            bk0.q[0] = *(const uint4*)(krow + hf * 8);
            bk0.q[1] = *(const uint4*)(krow + hf * 8 + 16);
            bk1.q[0] = *(const uint4*)(krow + 32 + hf * 8);
            bk1.q[1] = *(const uint4*)(krow + 32 + hf * 8 + 16);
            v8f a;
#pragma unroll
            for (int e = 0; e < 8; ++e) a[e] = 0.0f;
            a = wmma_bf(aq0, bk0, a);
            a = wmma_bf(aq1, bk1, a);
            S[t] = a;
            madd[t] = amask[bb * LKV + nk + t * 16 + l15] ? 0.0f : -1.0e30f;
        }
#pragma unroll
        for (int t = 0; t < 4; ++t)
#pragma unroll
            for (int e = 0; e < 8; ++e)
                S[t][e] = S[t][e] * 0.125f + madd[t];   // 1/sqrt(64) + mask

        // ---- online softmax (row reductions across 16-lane halves) ----
#pragma unroll
        for (int e = 0; e < 8; ++e) {
            float mx = fmaxf(fmaxf(S[0][e], S[1][e]), fmaxf(S[2][e], S[3][e]));
#pragma unroll
            for (int d = 1; d < 16; d <<= 1) mx = fmaxf(mx, __shfl_xor(mx, d, 32));
            const float mnew  = fmaxf(mrow[e], mx);
            const float alpha = __expf(mrow[e] - mnew);
            float rs = 0.0f;
#pragma unroll
            for (int t = 0; t < 4; ++t) {
                const float p = __expf(S[t][e] - mnew);
                S[t][e] = p; rs += p;
            }
#pragma unroll
            for (int d = 1; d < 16; d <<= 1) rs += __shfl_xor(rs, d, 32);
            lrow[e] = lrow[e] * alpha + rs;
            mrow[e] = mnew;
#pragma unroll
            for (int od = 0; od < 4; ++od) O[od][e] *= alpha;
        }

        // ---- relayout P: C-layout -> A-layout via per-wave LDS ----
#pragma unroll
        for (int e = 0; e < 8; ++e)
#pragma unroll
            for (int t = 0; t < 4; ++t)
                pbuf[wave][hf * 8 + e][t * 16 + l15] = f2b(S[t][e]);

        FragBf pa0, pa1;
        {
            const u16* prow = &pbuf[wave][l15][0];
            const int kb = hf * 8;
            pa0.q[0] = *(const uint4*)(prow + kb);
            pa0.q[1] = *(const uint4*)(prow + kb + 16);
            pa1.q[0] = *(const uint4*)(prow + 32 + kb);
            pa1.q[1] = *(const uint4*)(prow + 32 + kb + 16);
        }

        // ---- O += P @ V (K-dim = kv chunk of 64) ----
#pragma unroll
        for (int od = 0; od < 4; ++od) {
            const u16* vrow = &lsV[buf][(od * 16 + l15) * 64];
            FragBf bv0, bv1;
            bv0.q[0] = *(const uint4*)(vrow + hf * 8);
            bv0.q[1] = *(const uint4*)(vrow + hf * 8 + 16);
            bv1.q[0] = *(const uint4*)(vrow + 32 + hf * 8);
            bv1.q[1] = *(const uint4*)(vrow + 32 + hf * 8 + 16);
            O[od] = wmma_bf(pa0, bv0, O[od]);
            O[od] = wmma_bf(pa1, bv1, O[od]);
        }

        __syncthreads();
        buf ^= 1;
    }

    // ---- normalize and store ctx (bf16) ----
#pragma unroll
    for (int e = 0; e < 8; ++e) {
        const float inv = 1.0f / lrow[e];
        const int gm = bb * LQ + m0 + hf * 8 + e;
#pragma unroll
        for (int od = 0; od < 4; ++od)
            ctx[(size_t)gm * DIMN + hh * DHN + od * 16 + l15] = f2b(O[od][e] * inv);
    }
}

// ---------------- host launcher ----------------
extern "C" void kernel_launch(void* const* d_in, const int* in_sizes, int n_in,
                              void* d_out, int out_size, void* d_ws, size_t ws_size,
                              hipStream_t stream)
{
    (void)in_sizes; (void)n_in; (void)out_size; (void)ws_size;
    const float* q    = (const float*)d_in[0];
    const float* kv   = (const float*)d_in[1];
    const unsigned char* amask = (const unsigned char*)d_in[2];  // bool array
    const float* ln1g = (const float*)d_in[3];
    const float* ln1b = (const float*)d_in[4];
    const float* Wq   = (const float*)d_in[5];
    const float* Wkv  = (const float*)d_in[6];
    const float* Wo   = (const float*)d_in[7];
    const float* ln2g = (const float*)d_in[8];
    const float* ln2b = (const float*)d_in[9];
    const float* W1   = (const float*)d_in[10];
    const float* Wg   = (const float*)d_in[11];
    const float* W2   = (const float*)d_in[12];
    const float* pos  = (const float*)d_in[13];
    float* out = (float*)d_out;

    size_t off = 0;
    auto alloc = [&](size_t bytes) -> void* {
        void* p = (char*)d_ws + off;
        off += (bytes + 255) & ~(size_t)255;
        return p;
    };
    u16*   xln   = (u16*)  alloc((size_t)MQ * DIMN * 2);       // LN1(q) bf16
    u16*   WqT   = (u16*)  alloc((size_t)DIMN * DIMN * 2);
    u16*   qb    = (u16*)  alloc((size_t)MQ * DIMN * 2);       // queries bf16
    u16*   kvpb  = (u16*)  alloc((size_t)MKV * DIMN * 2);      // kv+pos bf16
    u16*   WkvT  = (u16*)  alloc((size_t)2 * DIMN * DIMN * 2);
    u16*   keysb = (u16*)  alloc((size_t)MKV * DIMN * 2);
    u16*   vtb   = (u16*)  alloc((size_t)BQ * HEADS * DHN * LKV * 2);
    u16*   ctxb  = (u16*)  alloc((size_t)MQ * DIMN * 2);
    u16*   WoT   = (u16*)  alloc((size_t)DIMN * DIMN * 2);
    float* attnf = (float*)alloc((size_t)MQ * DIMN * 4);       // attn_out f32
    u16*   attnb = (u16*)  alloc((size_t)MQ * DIMN * 2);
    u16*   hb    = (u16*)  alloc((size_t)MQ * DIMN * 2);       // LN2 bf16
    u16*   W1T   = (u16*)  alloc((size_t)INNERN * DIMN * 2);
    u16*   WgT   = (u16*)  alloc((size_t)INNERN * DIMN * 2);
    u16*   g1b   = (u16*)  alloc((size_t)MQ * INNERN * 2);     // gelu(h@W1) bf16
    u16*   ffb   = (u16*)  alloc((size_t)MQ * INNERN * 2);     // GEGLU product bf16
    u16*   W2T   = (u16*)  alloc((size_t)DIMN * INNERN * 2);

    const dim3 tb(32, 8);
    transpose_to_bf16<<<dim3(DIMN / 32, DIMN / 32), tb, 0, stream>>>(Wq,  WqT,  DIMN, DIMN);
    transpose_to_bf16<<<dim3(2 * DIMN / 32, DIMN / 32), tb, 0, stream>>>(Wkv, WkvT, DIMN, 2 * DIMN);
    transpose_to_bf16<<<dim3(DIMN / 32, DIMN / 32), tb, 0, stream>>>(Wo,  WoT,  DIMN, DIMN);
    transpose_to_bf16<<<dim3(INNERN / 32, DIMN / 32), tb, 0, stream>>>(W1, W1T, DIMN, INNERN);
    transpose_to_bf16<<<dim3(INNERN / 32, DIMN / 32), tb, 0, stream>>>(Wg, WgT, DIMN, INNERN);
    transpose_to_bf16<<<dim3(DIMN / 32, INNERN / 32), tb, 0, stream>>>(W2, W2T, INNERN, DIMN);

    layernorm_to_bf16<<<MQ, 256, 0, stream>>>(q, ln1g, ln1b, xln);
    kv_plus_pos_bf16<<<(MKV * DIMN) / 1024, 256, 0, stream>>>(kv, pos, kvpb);

    // queries = LN1(q) @ Wq  -> bf16
    gemm_bf16<1><<<dim3(DIMN / 128, MQ / 256), 256, 0, stream>>>(
        xln, DIMN, WqT, DIMN, nullptr, qb, nullptr, nullptr, nullptr, MQ, DIMN, DIMN);
    // [keys|values] = kvp @ Wkv, split epilogue (V stored transposed per head)
    gemm_bf16<6><<<dim3(2 * DIMN / 128, MKV / 256), 256, 0, stream>>>(
        kvpb, DIMN, WkvT, DIMN, nullptr, keysb, vtb, nullptr, nullptr, MKV, 2 * DIMN, DIMN);
    // fused attention -> ctx bf16
    flash_attn<<<dim3(LQ / 128, HEADS, BQ), 256, 0, stream>>>(qb, keysb, vtb, amask, ctxb);
    // attn_out = ctx @ Wo + q  (f32 + bf16 copies)
    gemm_bf16<2><<<dim3(DIMN / 128, MQ / 256), 256, 0, stream>>>(
        ctxb, DIMN, WoT, DIMN, attnf, attnb, nullptr, q, nullptr, MQ, DIMN, DIMN);
    layernorm_to_bf16<<<MQ, 256, 0, stream>>>(attnf, ln2g, ln2b, hb);
    // g1 = gelu(h @ W1)
    gemm_bf16<3><<<dim3(INNERN / 128, MQ / 256), 256, 0, stream>>>(
        hb, DIMN, W1T, DIMN, nullptr, g1b, nullptr, nullptr, nullptr, MQ, INNERN, DIMN);
    // ff = g1 * (attn_out @ Wg)
    gemm_bf16<4><<<dim3(INNERN / 128, MQ / 256), 256, 0, stream>>>(
        attnb, DIMN, WgT, DIMN, nullptr, ffb, nullptr, nullptr, g1b, MQ, INNERN, DIMN);
    // out = ff @ W2 + attn_out   (f32)
    gemm_bf16<5><<<dim3(DIMN / 128, MQ / 256), 256, 0, stream>>>(
        ffb, INNERN, W2T, INNERN, out, nullptr, nullptr, attnf, nullptr, MQ, DIMN, INNERN);
}
